// DMPNNEncoder_35201551958459
// MI455X (gfx1250) — matrix operand order repaired and checked by reading, not compiled, not run on hardware
//
#include <hip/hip_runtime.h>
#include <hip/hip_bf16.h>

// ---------------------------------------------------------------------------
// D-MPNN encoder for MI455X (gfx1250), wave32 + WMMA bf16 (fp32 accumulate).
//
// Algebraic restructuring:
//   edge GEMM (A[dst]-self*msg)@Wh^T  ->  node GEMM B = A@Wh^T + bh, gathered
//   per edge, with exact per-edge matvec correction only on self-loop edges.
// Fusion:
//   messages are never materialized per edge; the update kernel accumulates
//   relu'd messages straight into the next depth's node accumulator (atomic
//   fp32, L2-resident).  Full message rows are kept only for self-loop edges
//   (compact ping-pong buffer, expected ~E/N ~ 10 rows).
// ---------------------------------------------------------------------------

typedef __bf16 bf16_t;
typedef __attribute__((ext_vector_type(16))) __bf16 v16bf;
typedef __attribute__((ext_vector_type(8)))  float  v8f;
typedef __attribute__((ext_vector_type(4)))  int    v4i;

#define HID      300   // hidden size
#define HP       304   // hidden padded to 19*16 (WMMA N tiles)
#define KIN      128   // 127 (NODE_F+EDGE_F) padded to 4*32
#define KH       320   // 300 padded to 10*32 (WMMA K steps)
#define NODE_F   117
#define EDGE_F   10
#define DEPTH    3
#define CG       (HP / 8)   // 38 column groups of 8
#define SELF_CAP 4096       // max tracked self-loop edges (expected ~E/N)

// ---- WMMA fragment loaders (CDNA5 16-bit layouts, CDNA5 ISA §7.12.2) -----

// A-matrix 16x32 bf16 from row-major LDS tile (ld elems/row).
// lane L: M = L&15, K-half = L>>4; 16 bf16 at k-offsets {kh*8.., kh*8+16..}.
__device__ __forceinline__ v16bf load_fragA_lds(const bf16_t* tile, int m0,
                                                int k0, int ld) {
  int lane = threadIdx.x & 31;
  int row  = m0 + (lane & 15);
  int kh   = lane >> 4;
  const bf16_t* p = tile + row * ld + k0 + kh * 8;
  v16bf f;
  ((v4i*)&f)[0] = *(const v4i*)(p);
  ((v4i*)&f)[1] = *(const v4i*)(p + 16);
  return f;
}

// B-matrix 32x16 bf16 from row-major global weight [Kpad, HP]:
// lane L: N = n0+(L&15), K-half = L>>4; 16 values at k-offsets {kh*8.., kh*8+16..}.
__device__ __forceinline__ v16bf load_fragB_glb(const bf16_t* W, int k0,
                                                int n0, int ld) {
  int lane = threadIdx.x & 31;
  int kh   = lane >> 4;
  const bf16_t* p = W + (size_t)(k0 + kh * 8) * ld + n0 + (lane & 15);
  v16bf f;
#pragma unroll
  for (int s = 0; s < 8; ++s) {
    f[s]     = p[(size_t)s * ld];
    f[s + 8] = p[(size_t)(s + 16) * ld];
  }
  return f;
}

// Warm L0/L2 with the weight panel while LDS staging is in flight
// (lowers to global_prefetch_b8).
__device__ __forceinline__ void prefetch_weights(const bf16_t* W, int elems) {
  for (int off = (threadIdx.x & 63) * 64; off < elems; off += 64 * 64)
    __builtin_prefetch(W + off, 0, 0);
}

// ---- weight prep: WT[k*HP+n] = W[n*inLd + colOff + k] (bf16, zero-padded) --
__global__ __launch_bounds__(256)
void k_make_wt(const float* __restrict__ W, bf16_t* __restrict__ WT,
               int inLd, int colOff, int kValid, int outDim, int Kpad) {
  int tid = blockIdx.x * 256 + threadIdx.x;
  int total = Kpad * HP;
  if (tid >= total) return;
  int k = tid / HP, n = tid % HP;
  float v = (k < kValid && n < outDim) ? W[(size_t)n * inLd + colOff + k] : 0.f;
  WT[tid] = (bf16_t)v;
}

// ---- utility zero kernels --------------------------------------------------
__global__ __launch_bounds__(256)
void k_zero_f32(float* p, long n) {
  long i = (long)blockIdx.x * 256 + threadIdx.x;
  if (i < n) p[i] = 0.f;
}
__global__ __launch_bounds__(256)
void k_zero_i32(int* p, long n) {
  long i = (long)blockIdx.x * 256 + threadIdx.x;
  if (i < n) p[i] = 0;
}

// ---- degree / self-loop slot / neighbor-mask setup -------------------------
__global__ __launch_bounds__(256)
void k_degree(const int* __restrict__ idx, int* __restrict__ deg, int E) {
  int e = blockIdx.x * 256 + threadIdx.x;
  if (e < E) atomicAdd(&deg[idx[e]], 1);
}
__global__ __launch_bounds__(256)
void k_edge_setup(const int* __restrict__ esrc, const int* __restrict__ edst,
                  const int* __restrict__ degsrc, unsigned char* __restrict__ hasnb,
                  int* __restrict__ slot, int* __restrict__ cnt, int E) {
  int e = blockIdx.x * 256 + threadIdx.x;
  if (e < E) {
    bool self = (esrc[e] == edst[e]);
    hasnb[e] = (degsrc[edst[e]] - (self ? 1 : 0)) > 0 ? 1 : 0;
    slot[e]  = self ? atomicAdd(cnt, 1) : -1;
  }
}

// ---- K1: input_msg[E,HP] = [atom[dst] ; edge_feat] @ Wi^T + bi (bf16 out) --
// 64 rows/block, 2 waves; each wave computes two M-tiles per B-fragment.
__global__ __launch_bounds__(64)
void k_input_msg(const float* __restrict__ atom, const float* __restrict__ edgef,
                 const int* __restrict__ edst, const bf16_t* __restrict__ WiT,
                 const float* __restrict__ bi, bf16_t* __restrict__ out, int E) {
  __shared__ __align__(16) bf16_t As[64 * KIN];
  __shared__ int dsts[64];
  int e0 = blockIdx.x * 64;
  int t  = threadIdx.x;
  prefetch_weights(WiT, KIN * HP);
  { int e = e0 + t; dsts[t] = (e < E) ? edst[e] : 0; }
  __syncthreads();
  for (int i = t; i < 64 * KIN; i += 64) {
    int r = i >> 7, c = i & 127;
    int e = e0 + r;
    float v = 0.f;
    if (e < E) {
      if (c < NODE_F)               v = atom[(size_t)dsts[r] * NODE_F + c];
      else if (c < NODE_F + EDGE_F) v = edgef[(size_t)e * EDGE_F + (c - NODE_F)];
    }
    As[i] = (bf16_t)v;
  }
  __syncthreads();
  int w = t >> 5, lane = t & 31;
  int m0 = w * 32;                      // this wave: tiles m0 and m0+16
  bool full = (e0 + 64 <= E);           // uniform fast path (E % 64 == 0)
  for (int nt = 0; nt < HP / 16; ++nt) {
    int n0 = nt * 16;
    v8f acc0 = {}, acc1 = {};
#pragma unroll
    for (int ks = 0; ks < KIN / 32; ++ks) {
      v16bf b  = load_fragB_glb(WiT, ks * 32, n0, HP);
      v16bf a0 = load_fragA_lds(As, m0,      ks * 32, KIN);
      v16bf a1 = load_fragA_lds(As, m0 + 16, ks * 32, KIN);
      acc0 = __builtin_amdgcn_wmma_f32_16x16x32_bf16(false, a0, false, b,
                                                     (short)0, acc0, false, false);
      acc1 = __builtin_amdgcn_wmma_f32_16x16x32_bf16(false, a1, false, b,
                                                     (short)0, acc1, false, false);
    }
    int col  = n0 + (lane & 15);
    int hi   = lane >> 4;
    float bb = (col < HID) ? bi[col] : 0.f;
    int base0 = e0 + m0 + hi * 8;
    if (full) {
#pragma unroll
      for (int r = 0; r < 8; ++r)
        out[(size_t)(base0 + r) * HP + col] = (bf16_t)(acc0[r] + bb);
#pragma unroll
      for (int r = 0; r < 8; ++r)
        out[(size_t)(base0 + 16 + r) * HP + col] = (bf16_t)(acc1[r] + bb);
    } else {
#pragma unroll
      for (int r = 0; r < 8; ++r) {
        int row = base0 + r;
        if (row < E) out[(size_t)row * HP + col] = (bf16_t)(acc0[r] + bb);
      }
#pragma unroll
      for (int r = 0; r < 8; ++r) {
        int row = base0 + 16 + r;
        if (row < E) out[(size_t)row * HP + col] = (bf16_t)(acc1[r] + bb);
      }
    }
  }
}

// ---- K2: B[N,HP] = A[N,HP] @ Wh^T + bh  (A fp32 -> bf16 LDS stage) --------
__global__ __launch_bounds__(64)
void k_node_gemm(const float* __restrict__ A, const bf16_t* __restrict__ WhT,
                 const float* __restrict__ bh, float* __restrict__ B, int Nn) {
  __shared__ __align__(16) bf16_t As[64 * KH];  // 40 KB
  int g0 = blockIdx.x * 64;
  int t  = threadIdx.x;
  prefetch_weights(WhT, KH * HP);
  for (int i = t; i < 64 * KH; i += 64) {
    int r = i / KH, c = i % KH;
    int n = g0 + r;
    float v = (n < Nn && c < HP) ? A[(size_t)n * HP + c] : 0.f;
    As[i] = (bf16_t)v;
  }
  __syncthreads();
  int w = t >> 5, lane = t & 31;
  int m0 = w * 32;
  bool full = (g0 + 64 <= Nn);
  for (int nt = 0; nt < HP / 16; ++nt) {
    int n0 = nt * 16;
    v8f acc0 = {}, acc1 = {};
#pragma unroll
    for (int ks = 0; ks < KH / 32; ++ks) {
      v16bf b  = load_fragB_glb(WhT, ks * 32, n0, HP);
      v16bf a0 = load_fragA_lds(As, m0,      ks * 32, KH);
      v16bf a1 = load_fragA_lds(As, m0 + 16, ks * 32, KH);
      acc0 = __builtin_amdgcn_wmma_f32_16x16x32_bf16(false, a0, false, b,
                                                     (short)0, acc0, false, false);
      acc1 = __builtin_amdgcn_wmma_f32_16x16x32_bf16(false, a1, false, b,
                                                     (short)0, acc1, false, false);
    }
    int col  = n0 + (lane & 15);
    int hi   = lane >> 4;
    float bb = (col < HID) ? bh[col] : 0.f;
    int base0 = g0 + m0 + hi * 8;
    if (full) {
#pragma unroll
      for (int r = 0; r < 8; ++r)
        B[(size_t)(base0 + r) * HP + col] = acc0[r] + bb;
#pragma unroll
      for (int r = 0; r < 8; ++r)
        B[(size_t)(base0 + 16 + r) * HP + col] = acc1[r] + bb;
    } else {
#pragma unroll
      for (int r = 0; r < 8; ++r) {
        int row = base0 + r;
        if (row < Nn) B[(size_t)row * HP + col] = acc0[r] + bb;
      }
#pragma unroll
      for (int r = 0; r < 8; ++r) {
        int row = base0 + 16 + r;
        if (row < Nn) B[(size_t)row * HP + col] = acc1[r] + bb;
      }
    }
  }
}

// ---- K3: fused edge update + scatter ---------------------------------------
// val = relu(inmsg + hasnb*(B[dst] - self*corr)); Aout[accIdx[e]] += val.
// Messages only materialized for self-loop edges (ping-pong side buffer).
__global__ __launch_bounds__(256)
void k_update_fused(const bf16_t* __restrict__ inmsg, const float* __restrict__ B,
                    const int* __restrict__ edst,
                    const unsigned char* __restrict__ hasnb,
                    const int* __restrict__ slot,
                    const bf16_t* __restrict__ selfOld, bf16_t* __restrict__ selfNew,
                    const int* __restrict__ accIdx, float* __restrict__ Aout,
                    const bf16_t* __restrict__ WhT, int E) {
  int tid = blockIdx.x * 256 + threadIdx.x;
  if (tid >= E * CG) return;
  int e = tid / CG, g = tid % CG;
  int c0 = g * 8;
  int dst = edst[e];
  bool hnb = hasnb[e] != 0;
  int sl = slot[e];  // >= 0 iff self-loop edge
  v4i mi = *(const v4i*)(inmsg + (size_t)e * HP + c0);
  const bf16_t* miv = (const bf16_t*)&mi;
  float vals[8];
#pragma unroll
  for (int i = 0; i < 8; ++i) vals[i] = (float)miv[i];
  if (hnb) {
    const float* brow = B + (size_t)dst * HP + c0;
    float bb[8];
#pragma unroll
    for (int i = 0; i < 8; ++i) bb[i] = brow[i];
    if (sl >= 0 && sl < SELF_CAP) {  // rare: subtract own message @ Wh^T
      const bf16_t* mo = selfOld + (size_t)sl * HP;
#pragma unroll
      for (int i = 0; i < 8; ++i) {
        float corr = 0.f;
        for (int k = 0; k < HID; ++k)
          corr += (float)mo[k] * (float)WhT[k * HP + c0 + i];
        bb[i] -= corr;
      }
    }
#pragma unroll
    for (int i = 0; i < 8; ++i) vals[i] += bb[i];
  }
#pragma unroll
  for (int i = 0; i < 8; ++i) vals[i] = vals[i] > 0.f ? vals[i] : 0.f;
  if (sl >= 0 && sl < SELF_CAP) {
    __align__(16) bf16_t tmp[8];
#pragma unroll
    for (int i = 0; i < 8; ++i) tmp[i] = (bf16_t)vals[i];
    *(v4i*)(selfNew + (size_t)sl * HP + c0) = *(v4i*)tmp;
  }
  float* arow = Aout + (size_t)accIdx[e] * HP + c0;
#pragma unroll
  for (int i = 0; i < 8; ++i) atomicAdd(arow + i, vals[i]);
}

// ---- K4: readout GEMM [atom ; in_agg] @ Wo^T + bo, relu, deg mask, col-sum -
__global__ __launch_bounds__(64)
void k_readout(const float* __restrict__ atom, const float* __restrict__ inagg,
               const int* __restrict__ degin, const bf16_t* __restrict__ WoTa,
               const bf16_t* __restrict__ WoTm, const float* __restrict__ bo,
               float* __restrict__ molsum, int Nn) {
  __shared__ __align__(16) bf16_t A1[64 * KIN];  // 16 KB
  __shared__ __align__(16) bf16_t A2[64 * KH];   // 40 KB
  __shared__ float mask[64];
  int g0 = blockIdx.x * 64;
  int t  = threadIdx.x;
  prefetch_weights(WoTm, KH * HP);
  { int n = g0 + t; mask[t] = (n < Nn && degin[n] > 0) ? 1.f : 0.f; }
  for (int i = t; i < 64 * KIN; i += 64) {
    int r = i >> 7, c = i & 127;
    int n = g0 + r;
    float v = (n < Nn && c < NODE_F) ? atom[(size_t)n * NODE_F + c] : 0.f;
    A1[i] = (bf16_t)v;
  }
  for (int i = t; i < 64 * KH; i += 64) {
    int r = i / KH, c = i % KH;
    int n = g0 + r;
    float v = (n < Nn && c < HP) ? inagg[(size_t)n * HP + c] : 0.f;
    A2[i] = (bf16_t)v;
  }
  __syncthreads();
  int w = t >> 5, lane = t & 31;
  int m0 = w * 32;
  for (int nt = 0; nt < HP / 16; ++nt) {
    int n0 = nt * 16;
    v8f acc0 = {}, acc1 = {};
#pragma unroll
    for (int ks = 0; ks < KIN / 32; ++ks) {
      v16bf b  = load_fragB_glb(WoTa, ks * 32, n0, HP);
      v16bf a0 = load_fragA_lds(A1, m0,      ks * 32, KIN);
      v16bf a1 = load_fragA_lds(A1, m0 + 16, ks * 32, KIN);
      acc0 = __builtin_amdgcn_wmma_f32_16x16x32_bf16(false, a0, false, b,
                                                     (short)0, acc0, false, false);
      acc1 = __builtin_amdgcn_wmma_f32_16x16x32_bf16(false, a1, false, b,
                                                     (short)0, acc1, false, false);
    }
#pragma unroll
    for (int ks = 0; ks < KH / 32; ++ks) {
      v16bf b  = load_fragB_glb(WoTm, ks * 32, n0, HP);
      v16bf a0 = load_fragA_lds(A2, m0,      ks * 32, KH);
      v16bf a1 = load_fragA_lds(A2, m0 + 16, ks * 32, KH);
      acc0 = __builtin_amdgcn_wmma_f32_16x16x32_bf16(false, a0, false, b,
                                                     (short)0, acc0, false, false);
      acc1 = __builtin_amdgcn_wmma_f32_16x16x32_bf16(false, a1, false, b,
                                                     (short)0, acc1, false, false);
    }
    int col  = n0 + (lane & 15);
    int hi   = lane >> 4;
    float bb = (col < HID) ? bo[col] : 0.f;
    float s  = 0.f;
#pragma unroll
    for (int r = 0; r < 8; ++r) {
      float v = acc0[r] + bb;
      v = (v > 0.f) ? v : 0.f;
      s += v * mask[m0 + hi * 8 + r];
    }
#pragma unroll
    for (int r = 0; r < 8; ++r) {
      float v = acc1[r] + bb;
      v = (v > 0.f) ? v : 0.f;
      s += v * mask[m0 + 16 + hi * 8 + r];
    }
    atomicAdd(&molsum[col], s);
  }
}

// ---- K5: out = relu((molsum/N) @ Wout^T + bout) ----------------------------
__global__ __launch_bounds__(320)
void k_final(const float* __restrict__ molsum, const float* __restrict__ Wout,
             const float* __restrict__ bout, float* __restrict__ out, int Nn) {
  __shared__ float mol[HID];
  int t = threadIdx.x;
  if (t < HID) mol[t] = molsum[t] / (float)Nn;
  __syncthreads();
  if (t < HID) {
    float s = bout[t];
    const float* wrow = Wout + (size_t)t * HID;
    for (int k = 0; k < HID; ++k) s += mol[k] * wrow[k];
    out[t] = (s > 0.f) ? s : 0.f;
  }
}

// ---------------------------------------------------------------------------
extern "C" void kernel_launch(void* const* d_in, const int* in_sizes, int n_in,
                              void* d_out, int out_size, void* d_ws, size_t ws_size,
                              hipStream_t stream) {
  const float* atom  = (const float*)d_in[0];
  const float* edgef = (const float*)d_in[1];
  const int*   esrc  = (const int*)d_in[2];
  const int*   edst  = (const int*)d_in[3];
  const float* Wi    = (const float*)d_in[4];
  const float* bi    = (const float*)d_in[5];
  const float* Wh    = (const float*)d_in[6];
  const float* bh    = (const float*)d_in[7];
  const float* Wo    = (const float*)d_in[8];
  const float* bo    = (const float*)d_in[9];
  const float* Wout  = (const float*)d_in[10];
  const float* bout  = (const float*)d_in[11];
  const int N = in_sizes[0] / NODE_F;
  const int E = in_sizes[2];
  float* out = (float*)d_out;

  // workspace carve (256 B aligned)
  char* ws = (char*)d_ws;
  auto carve = [&](size_t bytes) {
    char* p = ws;
    ws += (bytes + 255) & ~(size_t)255;
    return p;
  };
  bf16_t* inmsg = (bf16_t*)carve((size_t)E * HP * sizeof(bf16_t));
  float*  A0    = (float*)carve((size_t)N * HP * sizeof(float));
  float*  A1    = (float*)carve((size_t)N * HP * sizeof(float));
  float*  Bn    = (float*)carve((size_t)N * HP * sizeof(float));
  bf16_t* WiT   = (bf16_t*)carve((size_t)KIN * HP * sizeof(bf16_t));
  bf16_t* WhT   = (bf16_t*)carve((size_t)KH * HP * sizeof(bf16_t));
  bf16_t* WoTa  = (bf16_t*)carve((size_t)KIN * HP * sizeof(bf16_t));
  bf16_t* WoTm  = (bf16_t*)carve((size_t)KH * HP * sizeof(bf16_t));
  bf16_t* S0    = (bf16_t*)carve((size_t)SELF_CAP * HP * sizeof(bf16_t));
  bf16_t* S1    = (bf16_t*)carve((size_t)SELF_CAP * HP * sizeof(bf16_t));
  int*    deg   = (int*)carve((size_t)N * sizeof(int));
  int*    slot  = (int*)carve((size_t)E * sizeof(int));
  unsigned char* hasnb = (unsigned char*)carve((size_t)E);
  int*    scnt  = (int*)carve(sizeof(int));
  float*  molsum = (float*)carve((size_t)HP * sizeof(float));

  const int  eB  = (E + 255) / 256;
  const int  nB  = (N + 255) / 256;
  const long nHP = (long)N * HP;

  // weights -> transposed, padded bf16
  k_make_wt<<<(KIN * HP + 255) / 256, 256, 0, stream>>>(Wi, WiT, 127, 0, 127, HID, KIN);
  k_make_wt<<<(KH  * HP + 255) / 256, 256, 0, stream>>>(Wh, WhT, 300, 0, 300, HID, KH);
  k_make_wt<<<(KIN * HP + 255) / 256, 256, 0, stream>>>(Wo, WoTa, 417, 0, 117, HID, KIN);
  k_make_wt<<<(KH  * HP + 255) / 256, 256, 0, stream>>>(Wo, WoTm, 417, 117, 300, HID, KH);

  // out-degree over src -> has_nb mask + self-loop slots
  k_zero_i32<<<nB, 256, 0, stream>>>(deg, N);
  k_zero_i32<<<1, 256, 0, stream>>>(scnt, 1);
  k_degree<<<eB, 256, 0, stream>>>(esrc, deg, E);
  k_edge_setup<<<eB, 256, 0, stream>>>(esrc, edst, deg, hasnb, slot, scnt, E);

  // input_msg (gather + WMMA GEMM)
  k_input_msg<<<(E + 63) / 64, 64, 0, stream>>>(atom, edgef, edst, WiT, bi, inmsg, E);

  // fresh state every call (harness does not re-poison between replays)
  k_zero_f32<<<(int)((nHP + 255) / 256), 256, 0, stream>>>(A0, nHP);
  k_zero_f32<<<(SELF_CAP * HP / 2 + 255) / 256, 256, 0, stream>>>((float*)S0,
                                                                 SELF_CAP * HP / 2);

  float*  Ain = A0;
  bf16_t* sOld = S0;
  const int updB = (E * CG + 255) / 256;
  for (int d = 0; d < DEPTH; ++d) {
    // B = A @ Wh^T + bh (node level: 10x fewer FLOPs than edge-level GEMM)
    k_node_gemm<<<(N + 63) / 64, 64, 0, stream>>>(Ain, WhT, bh, Bn, N);
    float*  Aout = (Ain == A0) ? A1 : A0;
    bf16_t* sNew = (sOld == S0) ? S1 : S0;
    k_zero_f32<<<(int)((nHP + 255) / 256), 256, 0, stream>>>(Aout, nHP);
    // last depth accumulates over edge_dst: that IS the readout's in_agg
    const int* aidx = (d == DEPTH - 1) ? edst : esrc;
    k_update_fused<<<updB, 256, 0, stream>>>(inmsg, Bn, edst, hasnb, slot,
                                             sOld, sNew, aidx, Aout, WhT, E);
    Ain = Aout;
    sOld = sNew;
  }
  // Ain now holds in_agg = segment_sum(messages, edge_dst)

  // in-degree for readout mask
  k_zero_i32<<<nB, 256, 0, stream>>>(deg, N);
  k_degree<<<eB, 256, 0, stream>>>(edst, deg, E);

  // readout GEMM fused with masked column-sum
  k_zero_f32<<<1, 256, 0, stream>>>(molsum, HP);
  k_readout<<<(N + 63) / 64, 64, 0, stream>>>(atom, Ain, deg, WoTa, WoTm, bo,
                                              molsum, N);

  // final matvec + relu
  k_final<<<1, 320, 0, stream>>>(molsum, Wout, bout, out, N);
}
